// Encoder_2001454760094
// MI455X (gfx1250) — matrix operand order repaired and checked by reading, not compiled, not run on hardware
//
#include <hip/hip_runtime.h>
#include <hip/hip_bf16.h>

// ---------------------------------------------------------------------------
// GCN forward: out = relu( D^-1/2 (A+I) D^-1/2 (x @ W) + bias )
//   N = 100000 nodes, F = 128 features, E = 1.6M edges
// Pipeline:
//   1) deg[i] = 1 (self loop) ; atomic += per dst edge ; dinv = rsqrt(deg)
//   2) h = x @ W    -- f16 WMMA (v_wmma_f32_16x16x32_f16), f32 accumulate
//   3) out[i] = dinv[i]^2 * h[i]            (self-loop term, inits d_out)
//   4) out[dst] += dinv[src]*dinv[dst]*h[src]  (f32 global atomics, 1 wave/edge)
//   5) out = relu(out + bias)
// ---------------------------------------------------------------------------

#define N_FEAT 128

typedef _Float16 v16h __attribute__((ext_vector_type(16)));
typedef _Float16 v8h  __attribute__((ext_vector_type(8)));
typedef float    v8f  __attribute__((ext_vector_type(8)));

// ---------------------------- degree kernels -------------------------------

__global__ void k_init_deg(float* __restrict__ deg, int n) {
    int i = blockIdx.x * blockDim.x + threadIdx.x;
    if (i < n) deg[i] = 1.0f;                 // self-loop contributes 1
}

__global__ void k_deg(const int* __restrict__ ei, float* __restrict__ deg,
                      int nedges, int nnodes) {
    int e = blockIdx.x * blockDim.x + threadIdx.x;
    if (e >= nedges) return;
    int d = ei[nedges + e];                   // row 1 = dst
    if ((unsigned)d < (unsigned)nnodes) atomicAdd(&deg[d], 1.0f);
}

__global__ void k_dinv(float* __restrict__ deg, int n) {
    int i = blockIdx.x * blockDim.x + threadIdx.x;
    if (i < n) deg[i] = rsqrtf(deg[i]);       // deg >= 1 always
}

// ------------------------------ WMMA GEMM ----------------------------------
// One block = 16 output rows (one M-tile), 8 waves = 8 N-tiles of 16.
// W^T staged to LDS as f16 (rows padded to 136 halves -> conflict-free b128).

#define LDS_PITCH 136   // 272 B row stride: 68 banks == 4 mod 64 -> no conflicts

__global__ __launch_bounds__(256) void k_gemm(const float* __restrict__ x,
                                              const float* __restrict__ w,
                                              float* __restrict__ hbuf) {
    __shared__ _Float16 WhT[128][LDS_PITCH];  // WhT[n][k] = W[k][n]
    __shared__ _Float16 Xh[16][LDS_PITCH];    // Xh[m][k]  = x[m0+m][k]

    const int t  = threadIdx.x;
    const int m0 = blockIdx.x * 16;           // 100000 = 6250*16, no tail

    for (int idx = t; idx < 128 * 128; idx += 256) {
        int k = idx >> 7, n = idx & 127;
        WhT[n][k] = (_Float16)w[idx];
    }
    for (int idx = t; idx < 16 * 128; idx += 256) {
        int r = idx >> 7, c = idx & 127;
        Xh[r][c] = (_Float16)x[(size_t)(m0 + r) * N_FEAT + c];
    }
    __syncthreads();

    const int wave = t >> 5;                  // N-tile id 0..7
    const int lane = t & 31;
    const int lmod = lane & 15;
    const int lhi  = lane >> 4;               // 0 / 1 (half-wave)
    const int n0   = wave * 16;

    // ISA layout (CDNA5 ISA 7.12.2):
    //  A lane L, elems e: M=L%16, K = (e&8)*2 + (L>=16?8:0) + (e&7)
    //     -> two contiguous 8-half chunks at K = klo and K = klo+16
    //  B lane L, elems e: N=L%16, K = (e&8)*2 + (L>=16?8:0) + (e&7)  (same as A)
    const int klo = lhi * 8;
    const int kb  = lhi * 16;

    v8f acc = {};
    #pragma unroll
    for (int kc = 0; kc < 4; ++kc) {          // K = 128 in chunks of 32
        const int kbase = kc * 32;
        union { v16h v; v8h q[2]; } A, B;
        A.q[0] = *(const v8h*)&Xh[lmod][kbase + klo];
        A.q[1] = *(const v8h*)&Xh[lmod][kbase + klo + 16];
        B.q[0] = *(const v8h*)&WhT[n0 + lmod][kbase + klo];
        B.q[1] = *(const v8h*)&WhT[n0 + lmod][kbase + klo + 16];
        acc = __builtin_amdgcn_wmma_f32_16x16x32_f16(
                  false, A.v, false, B.v, (short)0, acc, false, false);
    }

    // D: elem r -> row M = r + lhi*8, col = n0 + lmod
    float* op = hbuf + (size_t)(m0 + lhi * 8) * N_FEAT + n0 + lmod;
    #pragma unroll
    for (int r = 0; r < 8; ++r)
        op[(size_t)r * N_FEAT] = acc[r];
}

// ----------------------- self-loop init of d_out ---------------------------

__global__ void k_selfloop(const float* __restrict__ hbuf,
                           const float* __restrict__ dinv,
                           float* __restrict__ out, int n) {
    size_t i = (size_t)blockIdx.x * blockDim.x + threadIdx.x; // over n*128
    if (i >= (size_t)n * N_FEAT) return;
    float di = dinv[i >> 7];
    out[i] = hbuf[i] * di * di;               // fully overwrites d_out
}

// ------------------------- edge scatter (hot loop) -------------------------
// 1 wave / edge; lane handles 4 contiguous features (float4 gather + 4 atomics)

__global__ __launch_bounds__(256) void k_scatter(const int* __restrict__ ei,
                                                 const float* __restrict__ hbuf,
                                                 const float* __restrict__ dinv,
                                                 float* __restrict__ out,
                                                 int nedges, int nnodes) {
    const int wave = (int)((blockIdx.x * (size_t)blockDim.x + threadIdx.x) >> 5);
    const int lane = threadIdx.x & 31;
    if (wave >= nedges) return;

    // prefetch a future edge's source row (global_prefetch_b8) to hide the
    // random-gather latency on h[src]
    const int PF = 2048;
    if (wave + PF < nedges) {
        int sp = ei[wave + PF];
        if ((unsigned)sp < (unsigned)nnodes)
            __builtin_prefetch(hbuf + (size_t)sp * N_FEAT + lane * 4, 0, 0);
    }

    int s = ei[wave];                         // row 0 = src
    int d = ei[nedges + wave];                // row 1 = dst
    if ((unsigned)s >= (unsigned)nnodes || (unsigned)d >= (unsigned)nnodes) return;

    float norm = dinv[s] * dinv[d];
    const float4 v = *(const float4*)(hbuf + (size_t)s * N_FEAT + lane * 4);
    float* o = out + (size_t)d * N_FEAT + lane * 4;
    atomicAdd(o + 0, norm * v.x);
    atomicAdd(o + 1, norm * v.y);
    atomicAdd(o + 2, norm * v.z);
    atomicAdd(o + 3, norm * v.w);
}

// ------------------------------ finalize -----------------------------------

__global__ void k_finalize(float* __restrict__ out,
                           const float* __restrict__ bias, int n) {
    size_t i = (size_t)blockIdx.x * blockDim.x + threadIdx.x;
    if (i >= (size_t)n * N_FEAT) return;
    float v = out[i] + bias[i & (N_FEAT - 1)];
    out[i] = v > 0.0f ? v : 0.0f;
}

// ------------------------------ launcher -----------------------------------

extern "C" void kernel_launch(void* const* d_in, const int* in_sizes, int n_in,
                              void* d_out, int out_size, void* d_ws, size_t ws_size,
                              hipStream_t stream) {
    const float* x    = (const float*)d_in[0];
    const int*   ei   = (const int*)d_in[1];
    const float* w    = (const float*)d_in[2];
    const float* bias = (const float*)d_in[3];
    float*       out  = (float*)d_out;

    const int nnodes = in_sizes[0] / N_FEAT;      // 100000
    const int nedges = in_sizes[1] / 2;           // 1600000

    // workspace layout: [deg/dinv : nnodes f32][h : nnodes*128 f32]
    float* deg  = (float*)d_ws;
    size_t degB = ((size_t)nnodes * sizeof(float) + 255) & ~(size_t)255;
    float* hbuf = (float*)((char*)d_ws + degB);

    const int B = 256;

    // 1) degrees -> dinv (in place)
    k_init_deg<<<(nnodes + B - 1) / B, B, 0, stream>>>(deg, nnodes);
    k_deg<<<(nedges + B - 1) / B, B, 0, stream>>>(ei, deg, nedges, nnodes);
    k_dinv<<<(nnodes + B - 1) / B, B, 0, stream>>>(deg, nnodes);

    // 2) h = x @ W via f16 WMMA (100000/16 = 6250 M-tiles)
    k_gemm<<<nnodes / 16, B, 0, stream>>>(x, w, hbuf);

    // 3) self-loop term initializes d_out
    size_t elems = (size_t)nnodes * N_FEAT;
    k_selfloop<<<(unsigned)((elems + B - 1) / B), B, 0, stream>>>(hbuf, deg, out, nnodes);

    // 4) edge scatter: 8 edges per 256-thread block (1 wave / edge)
    unsigned sblocks = (unsigned)((nedges + 7) / 8);
    k_scatter<<<sblocks, B, 0, stream>>>(ei, hbuf, deg, out, nedges, nnodes);

    // 5) bias + relu
    k_finalize<<<(unsigned)((elems + B - 1) / B), B, 0, stream>>>(out, bias, nnodes);
}
